// ResidualBlock_2001454760194
// MI455X (gfx1250) — matrix operand order repaired and checked by reading, not compiled, not run on hardware
//
#include <hip/hip_runtime.h>
#include <stdint.h>

#define N_PTS   400000
#define N_PAIRS 200000
#define CCH     128
#define K_VOL   27
#define GROUPS  8
#define CPG     16          // channels per group
#define TILE_M  128
#define NBLK    ((N_PAIRS + TILE_M - 1) / TILE_M)   // 1563 (last block partial)
#define NEG_SLOPE 0.01f
#define EPSV      1e-5f

typedef __bf16 bf16_t;
typedef __attribute__((ext_vector_type(16))) __bf16 v16bf;
typedef __attribute__((ext_vector_type(8)))  float  v8f;
typedef __attribute__((ext_vector_type(4)))  uint32_t v4u;
typedef __attribute__((ext_vector_type(8)))  uint32_t v8u;

// ---------------- elementwise helpers ----------------

__global__ __launch_bounds__(256) void zero_f4_kernel(float4* __restrict__ p, int n4) {
    int i = blockIdx.x * 256 + threadIdx.x;
    if (i < n4) p[i] = make_float4(0.f, 0.f, 0.f, 0.f);
}

__global__ void zero_f32_kernel(float* __restrict__ p, int n) {
    int i = blockIdx.x * blockDim.x + threadIdx.x;
    if (i < n) p[i] = 0.f;
}

// f32 -> bf16, 8 elements per thread (16B stores)
__global__ __launch_bounds__(256) void f32_to_bf16_kernel(const float* __restrict__ in,
                                                          bf16_t* __restrict__ out, int n8) {
    int i = blockIdx.x * 256 + threadIdx.x;
    if (i >= n8) return;
    const float4* p = ((const float4*)in) + (size_t)i * 2;
    float4 a = p[0], b = p[1];
    union { bf16_t h[8]; uint4 u; } o;
    o.h[0] = (bf16_t)a.x; o.h[1] = (bf16_t)a.y; o.h[2] = (bf16_t)a.z; o.h[3] = (bf16_t)a.w;
    o.h[4] = (bf16_t)b.x; o.h[5] = (bf16_t)b.y; o.h[6] = (bf16_t)b.z; o.h[7] = (bf16_t)b.w;
    ((uint4*)out)[i] = o.u;
}

// w[k][cin][cout] f32 -> wt[k][cout][cin] bf16 (transpose so WMMA B frags are contiguous)
__global__ __launch_bounds__(256) void wtrans_kernel(const float* __restrict__ w,
                                                     bf16_t* __restrict__ wt) {
    int k = blockIdx.x;
    const float* wk = w + (size_t)k * CCH * CCH;
    bf16_t*      wo = wt + (size_t)k * CCH * CCH;
    for (int i = threadIdx.x; i < CCH * CCH; i += 256) {
        int co = i >> 7, ci = i & 127;
        wo[i] = (bf16_t)wk[ci * CCH + co];
    }
}

// ---------------- sparse conv: async gather -> WMMA GEMM -> scatter-add ----------------

__global__ __launch_bounds__(256) void spconv_wmma_kernel(
    const bf16_t* __restrict__ src,   // [N_PTS][C] bf16 (gather source)
    const bf16_t* __restrict__ wt,    // [K_VOL][Cout][Cin] bf16
    const int*    __restrict__ kin,   // [K_VOL][N_PAIRS]
    const int*    __restrict__ kout,  // [K_VOL][N_PAIRS]
    float*        __restrict__ dst)   // [N_PTS][C] f32 accumulator (pre-zeroed)
{
    // LDS: padded rows to avoid 64-bank conflicts on fragment reads
    __shared__ int    s_in[TILE_M];
    __shared__ int    s_out[TILE_M];
    __shared__ bf16_t sA[TILE_M][136];   // 272 B row stride
    __shared__ bf16_t sW[CCH][144];      // 288 B row stride

    const int tid  = threadIdx.x;
    const int k    = blockIdx.y;
    const int base = blockIdx.x * TILE_M;

    if (tid < TILE_M) {
        int p = base + tid;
        s_in[tid] = (p < N_PAIRS) ? kin[k * N_PAIRS + p] : 0;
    } else {
        int p = base + tid - TILE_M;
        s_out[tid - TILE_M] = (p < N_PAIRS) ? kout[k * N_PAIRS + p] : 0;
    }

    // ---- TDM: DMA the 128x128 bf16 weight tile (as 64 x 4B elems per row, 128 rows)
    // into LDS with hardware padding: 32 B pad every 256 B -> 288 B LDS row stride.
    if (tid < 32) {
        uint64_t ga   = (uint64_t)(uintptr_t)(wt + (size_t)k * CCH * CCH);
        uint32_t ldsW = (uint32_t)(uintptr_t)(&sW[0][0]);
        v4u g0;
        g0.x = 1u;                                     // count=1, user mode, no gather
        g0.y = ldsW;                                   // lds_addr
        g0.z = (uint32_t)ga;                           // global_addr[31:0]
        g0.w = ((uint32_t)(ga >> 32) & 0x01FFFFFFu)    // global_addr[56:32]
             | (2u << 30);                             // type=2 (image)
        v8u g1;
        g1[0] = (2u << 16)      // data_size = 4 bytes
              | (1u << 20)      // pad_enable
              | (5u << 22)      // pad_interval: 64 DWORDs = 256 B
              | (7u << 25);     // pad_amount: 8 DWORDs = 32 B
        g1[1] = (64u << 16);    // atomic_barrier_addr=0 | tensor_dim0[15:0]=64
        g1[2] = (128u << 16);   // tensor_dim0[31:16]=0  | tensor_dim1[15:0]=128
        g1[3] = (64u << 16);    // tensor_dim1[31:16]=0  | tile_dim0=64
        g1[4] = 128u;           // tile_dim1=128 | tile_dim2=0
        g1[5] = 64u;            // tensor_dim0_stride[31:0]=64 elems
        g1[6] = 0u;             // stride0 hi | stride1 lo
        g1[7] = 0u;             // stride1 hi
        asm volatile("tensor_load_to_lds %0, %1" :: "s"(g0), "s"(g1) : "memory");
    }

    __syncthreads();   // s_in visible for gather address generation

    // ---- async gather: 128 rows x 16 x 16B chunks, straight into LDS (no VGPR round-trip)
    {
        const uint32_t ldsA = (uint32_t)(uintptr_t)(&sA[0][0]);
        for (int i = tid; i < TILE_M * (CCH / 8); i += 256) {
            int r = i >> 4, s = i & 15;
            uint32_t go = (uint32_t)s_in[r] * (CCH * 2u) + (uint32_t)s * 16u; // byte offset
            uint32_t la = ldsA + (uint32_t)r * 272u + (uint32_t)s * 16u;
            asm volatile("global_load_async_to_lds_b128 %0, %1, %2"
                         :: "v"(la), "v"(go), "s"(src) : "memory");
        }
        asm volatile("s_wait_asynccnt 0x0" ::: "memory");
    }
    if (tid < 32) {
        __builtin_amdgcn_s_wait_tensorcnt(0);   // TDM weight tile landed
    }
    __syncthreads();

    const int wave = tid >> 5;
    const int lane = tid & 31;
    const int g    = lane >> 4;      // lane half
    const int ln   = lane & 15;
    const int nt   = wave;           // 8 waves cover 8 Cout tiles
    const int col  = nt * 16 + ln;

    // B fragments (fixed per wave): B[kk][n] = Wt[n][kk]; element e -> K = kc*32 + g*16 + e
    union { v16bf v; uint4 u[2]; } bfr[4];
#pragma unroll
    for (int kc = 0; kc < 4; ++kc) {
        const bf16_t* brow = &sW[col][kc * 32 + g * 16];
        bfr[kc].u[0] = *(const uint4*)(brow);
        bfr[kc].u[1] = *(const uint4*)(brow + 8);
    }

#pragma unroll
    for (int mt = 0; mt < TILE_M / 16; ++mt) {
        v8f acc = {0.f, 0.f, 0.f, 0.f, 0.f, 0.f, 0.f, 0.f};
#pragma unroll
        for (int kc = 0; kc < 4; ++kc) {
            // A element e -> K = kc*32 + g*16 + e
            union { v16bf v; uint4 u[2]; } afr;
            const bf16_t* arow = &sA[mt * 16 + ln][kc * 32 + g * 16];
            afr.u[0] = *(const uint4*)(arow);
            afr.u[1] = *(const uint4*)(arow + 8);
            acc = __builtin_amdgcn_wmma_f32_16x16x32_bf16(
                false, afr.v, false, bfr[kc].v, (short)0, acc, false, false);
        }
        // D: VGPR e holds M = e (lanes 0-15) / e+8 (lanes 16-31)
#pragma unroll
        for (int e = 0; e < 8; ++e) {
            int mrow = mt * 16 + e + 8 * g;
            if (base + mrow < N_PAIRS) {
                int orow = s_out[mrow];
                atomicAdd(dst + (size_t)orow * CCH + col, acc[e]);
            }
        }
    }
}

// ---------------- group norm ----------------

// partial (sum, sumsq) per group; each block covers 128 rows
__global__ __launch_bounds__(256) void gn_stats_kernel(const float* __restrict__ h,
                                                       float* __restrict__ stats) {
    __shared__ float s1[256], s2[256];
    const int c  = threadIdx.x & 127;
    const int hh = threadIdx.x >> 7;
    const size_t row0 = (size_t)blockIdx.x * 128 + (size_t)hh * 64;
    float sum = 0.f, sq = 0.f;
#pragma unroll 4
    for (int i = 0; i < 64; ++i) {
        float x = h[(row0 + i) * CCH + c];
        sum += x; sq += x * x;
    }
    s1[threadIdx.x] = sum; s2[threadIdx.x] = sq;
    __syncthreads();
    if (threadIdx.x < GROUPS) {
        int gr = threadIdx.x;
        float a = 0.f, b = 0.f;
        for (int j = 0; j < CPG; ++j) {
            int i0 = gr * CPG + j;
            a += s1[i0] + s1[128 + i0];
            b += s2[i0] + s2[128 + i0];
        }
        atomicAdd(&stats[gr], a);
        atomicAdd(&stats[GROUPS + gr], b);
    }
}

__global__ void gn_finalize_kernel(const float* __restrict__ stats,
                                   float* __restrict__ murstd) {
    int gr = threadIdx.x;
    if (gr >= GROUPS) return;
    const float cnt = (float)N_PTS * (float)CPG;
    float mu  = stats[gr] / cnt;
    float var = stats[GROUPS + gr] / cnt - mu * mu;
    murstd[gr]          = mu;
    murstd[GROUPS + gr] = rsqrtf(var + EPSV);
}

// normalize h (f32) -> bf16 output (gather source for conv2); 4 channels / thread
__global__ __launch_bounds__(256) void gn_apply_bf16_kernel(
    const float* __restrict__ h, const float* __restrict__ murstd,
    const float* __restrict__ gamma, const float* __restrict__ beta,
    bf16_t* __restrict__ out) {
    int i = blockIdx.x * 256 + threadIdx.x;          // handles 4 consecutive channels
    int c = (i * 4) & 127;
    int gr = c >> 4;
    float mu = murstd[gr], rs = murstd[GROUPS + gr];
    float4 x = ((const float4*)h)[i];
    union { bf16_t hh[4]; uint2 u; } o;
    o.hh[0] = (bf16_t)((x.x - mu) * rs * gamma[c + 0] + beta[c + 0]);
    o.hh[1] = (bf16_t)((x.y - mu) * rs * gamma[c + 1] + beta[c + 1]);
    o.hh[2] = (bf16_t)((x.z - mu) * rs * gamma[c + 2] + beta[c + 2]);
    o.hh[3] = (bf16_t)((x.w - mu) * rs * gamma[c + 3] + beta[c + 3]);
    ((uint2*)out)[i] = o.u;
}

// final: gn2(h2) + feats, leaky relu, f32 out
__global__ __launch_bounds__(256) void gn_residual_lrelu_kernel(
    const float* __restrict__ h, const float* __restrict__ murstd,
    const float* __restrict__ gamma, const float* __restrict__ beta,
    const float* __restrict__ feats, float* __restrict__ out) {
    int i = blockIdx.x * 256 + threadIdx.x;
    int c = (i * 4) & 127;
    int gr = c >> 4;
    float mu = murstd[gr], rs = murstd[GROUPS + gr];
    float4 x = ((const float4*)h)[i];
    float4 f = ((const float4*)feats)[i];
    float4 y;
    y.x = (x.x - mu) * rs * gamma[c + 0] + beta[c + 0] + f.x;
    y.y = (x.y - mu) * rs * gamma[c + 1] + beta[c + 1] + f.y;
    y.z = (x.z - mu) * rs * gamma[c + 2] + beta[c + 2] + f.z;
    y.w = (x.w - mu) * rs * gamma[c + 3] + beta[c + 3] + f.w;
    y.x = y.x > 0.f ? y.x : NEG_SLOPE * y.x;
    y.y = y.y > 0.f ? y.y : NEG_SLOPE * y.y;
    y.z = y.z > 0.f ? y.z : NEG_SLOPE * y.z;
    y.w = y.w > 0.f ? y.w : NEG_SLOPE * y.w;
    ((float4*)out)[i] = y;
}

// ---------------- launch ----------------

extern "C" void kernel_launch(void* const* d_in, const int* in_sizes, int n_in,
                              void* d_out, int out_size, void* d_ws, size_t ws_size,
                              hipStream_t stream) {
    (void)in_sizes; (void)n_in; (void)out_size; (void)ws_size;

    const float* feats  = (const float*)d_in[0];
    const float* w1     = (const float*)d_in[1];
    const float* gamma1 = (const float*)d_in[2];
    const float* beta1  = (const float*)d_in[3];
    const float* w2     = (const float*)d_in[4];
    const float* gamma2 = (const float*)d_in[5];
    const float* beta2  = (const float*)d_in[6];
    const int*   kin    = (const int*)d_in[7];
    const int*   kout   = (const int*)d_in[8];
    float*       out    = (float*)d_out;

    char* ws = (char*)d_ws;
    size_t off = 0;
    auto alloc = [&](size_t bytes) -> void* {
        void* p = ws + off;
        off += (bytes + 255) & ~(size_t)255;
        return p;
    };
    const size_t NE = (size_t)N_PTS * CCH;           // 51.2M elements
    bf16_t* featsbf = (bf16_t*)alloc(NE * sizeof(bf16_t));            // 102.4 MB
    float*  hbuf    = (float*) alloc(NE * sizeof(float));             // 204.8 MB (h1, reused as h2)
    bf16_t* h1nbf   = (bf16_t*)alloc(NE * sizeof(bf16_t));            // 102.4 MB
    bf16_t* wt1     = (bf16_t*)alloc((size_t)K_VOL * CCH * CCH * 2);  // 0.88 MB
    bf16_t* wt2     = (bf16_t*)alloc((size_t)K_VOL * CCH * CCH * 2);
    float*  stats1  = (float*) alloc(2 * GROUPS * sizeof(float));
    float*  murstd1 = (float*) alloc(2 * GROUPS * sizeof(float));
    float*  stats2  = (float*) alloc(2 * GROUPS * sizeof(float));
    float*  murstd2 = (float*) alloc(2 * GROUPS * sizeof(float));

    const int n4 = (int)(NE / 4);    // 12.8M float4
    const int g4 = n4 / 256;         // 50000 blocks
    const int n8 = (int)(NE / 8);    // 6.4M
    const int g8 = n8 / 256;         // 25000 blocks
    const dim3 gconv(NBLK, K_VOL);   // 1563 x 27

    zero_f32_kernel<<<1, 64, 0, stream>>>(stats1, 2 * GROUPS);
    zero_f32_kernel<<<1, 64, 0, stream>>>(stats2, 2 * GROUPS);
    // precision transforms (once per launch, cheap)
    f32_to_bf16_kernel<<<g8, 256, 0, stream>>>(feats, featsbf, n8);
    wtrans_kernel<<<K_VOL, 256, 0, stream>>>(w1, wt1);
    wtrans_kernel<<<K_VOL, 256, 0, stream>>>(w2, wt2);
    // conv1
    zero_f4_kernel<<<g4, 256, 0, stream>>>((float4*)hbuf, n4);
    spconv_wmma_kernel<<<gconv, 256, 0, stream>>>(featsbf, wt1, kin, kout, hbuf);
    // group norm 1 -> bf16 gather source for conv2
    gn_stats_kernel<<<N_PTS / 128, 256, 0, stream>>>(hbuf, stats1);
    gn_finalize_kernel<<<1, GROUPS, 0, stream>>>(stats1, murstd1);
    gn_apply_bf16_kernel<<<g4, 256, 0, stream>>>(hbuf, murstd1, gamma1, beta1, h1nbf);
    // conv2 (reuse hbuf; stream order guarantees norm1 finished reading)
    zero_f4_kernel<<<g4, 256, 0, stream>>>((float4*)hbuf, n4);
    spconv_wmma_kernel<<<gconv, 256, 0, stream>>>(h1nbf, wt2, kin, kout, hbuf);
    // group norm 2 + residual + leaky relu -> f32 out
    gn_stats_kernel<<<N_PTS / 128, 256, 0, stream>>>(hbuf, stats2);
    gn_finalize_kernel<<<1, GROUPS, 0, stream>>>(stats2, murstd2);
    gn_residual_lrelu_kernel<<<g4, 256, 0, stream>>>(hbuf, murstd2, gamma2, beta2, feats, out);
}
